// AdamsGCN_10557029614291
// MI455X (gfx1250) — matrix operand order repaired and checked by reading, not compiled, not run on hardware
//
#include <hip/hip_runtime.h>

#define NNODES   100000
#define NEDGES   1600000
#define FIN      128
#define HID      64
#define NGRAPHS  128

typedef __attribute__((ext_vector_type(16))) _Float16 v16h;
typedef __attribute__((ext_vector_type(8)))  float    v8f;

union FragU { v16h h; unsigned int u[8]; };

__device__ __forceinline__ unsigned short h2us(_Float16 h) {
    union { _Float16 h; unsigned short u; } x; x.h = h; return x.u;
}

// Pack two consecutive A elements into one dword of f16x2 (overloads pick source dtype)
__device__ __forceinline__ unsigned int load_pack(const float* base, size_t off) {
    _Float16 a0 = (_Float16)base[off];
    _Float16 a1 = (_Float16)base[off + 1];
    return (unsigned int)h2us(a0) | ((unsigned int)h2us(a1) << 16);
}
__device__ __forceinline__ unsigned int load_pack(const _Float16* base, size_t off) {
    return *(const unsigned int*)(base + off);   // off is even -> 4B aligned
}

// ---------------- degree ----------------
__global__ void init_deg_kernel(float* deg, int n) {
    int i = blockIdx.x * blockDim.x + threadIdx.x;
    if (i < n) deg[i] = 1.0f;                    // self-loop
}
__global__ void count_deg_kernel(const int* __restrict__ dst, float* deg, int nE) {
    int i = blockIdx.x * blockDim.x + threadIdx.x;
    if (i < nE) atomicAdd(&deg[dst[i]], 1.0f);
}
__global__ void dinv_kernel(const float* __restrict__ deg, float* dinv, int n) {
    int i = blockIdx.x * blockDim.x + threadIdx.x;
    if (i < n) dinv[i] = rsqrtf(deg[i]);
}

// ---------------- WMMA GEMM: out[M,64] = A[M,K] @ W[K,64] ----------------
// 256 threads = 8 waves; block covers 128 rows; wave w -> rows [128*bx + 16w, +16)
template<int K, typename TA>
__global__ __launch_bounds__(256)
void gemm_wmma_kernel(const TA* __restrict__ A, const float* __restrict__ W,
                      float* __restrict__ out, int M) {
    __shared__ unsigned int   ldsA[128 * (K / 2)];   // 128 rows x K halfs, packed f16x2
    __shared__ unsigned short ldsB[HID * K];         // W transposed: Wt[n][k] as f16

    const int tid  = threadIdx.x;
    const int wave = tid >> 5;
    const int lane = tid & 31;
    const int rowBase = blockIdx.x * 128;

    // Stage W (f32 global, row-major [K][64]) -> LDS transposed f16 [64][K]
    for (int e = tid; e < HID * K; e += 256) {
        int k = e / HID, n = e % HID;
        ldsB[n * K + k] = h2us((_Float16)W[k * HID + n]);
    }
    // Stage A tile (convert to f16 pairs)
    for (int d = tid; d < 128 * (K / 2); d += 256) {
        int r  = d / (K / 2);
        int c2 = d % (K / 2);
        int row = rowBase + r;
        unsigned int val = 0u;
        if (row < M) val = load_pack(A, (size_t)row * K + 2 * c2);
        ldsA[d] = val;
    }
    __syncthreads();

    const int m  = lane & 15;     // M (for A) / N (for B) index within tile
    const int kh = lane >> 4;     // lane-half selector
    const unsigned int* ldsBu = (const unsigned int*)ldsB;

    v8f acc[4] = {};              // 16x16 tiles for N blocks 0..3 (HID=64)

    #pragma unroll
    for (int s = 0; s < K / 32; ++s) {
        FragU a;
        #pragma unroll
        for (int v = 0; v < 8; ++v) {
            // ISA 7.12.2: 16-bit A 16x32: koff = (v/4)*16 + half*8 + (v%4)*2
            int ko = 32 * s + ((v >> 2) << 4) + (kh << 3) + ((v & 3) << 1);
            a.u[v] = ldsA[(16 * wave + m) * (K / 2) + (ko >> 1)];
        }
        #pragma unroll
        for (int c = 0; c < 4; ++c) {
            FragU b;
            #pragma unroll
            for (int v = 0; v < 8; ++v) {
                // 16-bit B KxN, same k order as A: koff = (v/4)*16 + half*8 + (v%4)*2
                int kv = 32 * s + ((v >> 2) << 4) + (kh << 3) + ((v & 3) << 1);
                b.u[v] = ldsBu[((16 * c + m) * K + kv) >> 1];
            }
            acc[c] = __builtin_amdgcn_wmma_f32_16x16x32_f16(
                false, a.h, false, b.h, (short)0, acc[c], false, false);
        }
    }

    // D layout: VGPR r, lane L -> M = (L>>4)*8 + r, N = L&15
    #pragma unroll
    for (int c = 0; c < 4; ++c) {
        #pragma unroll
        for (int r = 0; r < 8; ++r) {
            int row = rowBase + 16 * wave + (kh << 3) + r;
            if (row < M) out[(size_t)row * HID + 16 * c + m] = acc[c][r];
        }
    }
}

// ---------------- edge scatter: agg[dst] += h[src] * dinv[src]*dinv[dst] ----------------
__global__ __launch_bounds__(256)
void scatter_kernel(const int* __restrict__ src, const int* __restrict__ dst,
                    const float* __restrict__ dinv, const float* __restrict__ h,
                    float* __restrict__ agg, int nE) {
    int e    = (blockIdx.x * blockDim.x + threadIdx.x) >> 5;   // wave per edge
    int lane = threadIdx.x & 31;
    if (e >= nE) return;
    int s = src[e], d = dst[e];
    float nrm = dinv[s] * dinv[d];
    const float* hs = h   + (size_t)s * HID;
    float*       ag = agg + (size_t)d * HID;
    atomicAdd(&ag[lane],      hs[lane]      * nrm);
    atomicAdd(&ag[lane + 32], hs[lane + 32] * nrm);
}

// ---------------- layer-1 epilogue: relu(agg + h/deg + b) -> f16 ----------------
__global__ void finish1_kernel(const float* __restrict__ agg, const float* __restrict__ h,
                               const float* __restrict__ deg, const float* __restrict__ b,
                               _Float16* __restrict__ out, int n) {
    int i = blockIdx.x * blockDim.x + threadIdx.x;
    if (i >= n * HID) return;
    int node = i / HID, f = i % HID;
    float v = agg[i] + h[i] * (1.0f / deg[node]) + b[f];
    out[i] = (_Float16)fmaxf(v, 0.0f);
}

// ---------------- layer-2 epilogue fused with mean-pool accumulation ----------------
__global__ void finish2_pool_kernel(const float* __restrict__ agg, const float* __restrict__ h,
                                    const float* __restrict__ deg, const float* __restrict__ b,
                                    const int* __restrict__ batch,
                                    float* __restrict__ pool, float* __restrict__ cnt, int n) {
    int i = blockIdx.x * blockDim.x + threadIdx.x;
    if (i >= n * HID) return;
    int node = i / HID, f = i % HID;
    float v = agg[i] + h[i] * (1.0f / deg[node]) + b[f];
    v = fmaxf(v, 0.0f);
    int g = batch[node];
    atomicAdd(&pool[g * HID + f], v);
    if (f == 0) atomicAdd(&cnt[g], 1.0f);
}

// ---------------- head: out[g] = mean_pool(g) @ W_out + b_out ----------------
__global__ void final_kernel(const float* __restrict__ pool, const float* __restrict__ cnt,
                             const float* __restrict__ Wout, const float* __restrict__ bout,
                             float* __restrict__ out) {
    int g = threadIdx.x;
    if (g >= NGRAPHS) return;
    float c = fmaxf(cnt[g], 1.0f);
    float s = 0.0f;
    #pragma unroll
    for (int f = 0; f < HID; ++f) s += (pool[g * HID + f] / c) * Wout[f];
    out[g] = s + bout[0];
}

extern "C" void kernel_launch(void* const* d_in, const int* in_sizes, int n_in,
                              void* d_out, int out_size, void* d_ws, size_t ws_size,
                              hipStream_t stream) {
    (void)in_sizes; (void)n_in; (void)out_size; (void)ws_size;

    const float* x     = (const float*)d_in[0];
    const int*   ei    = (const int*)  d_in[1];   // [2, E]: row0 = src, row1 = dst
    const int*   batch = (const int*)  d_in[2];
    const float* W1    = (const float*)d_in[3];
    const float* b1    = (const float*)d_in[4];
    const float* W2    = (const float*)d_in[5];
    const float* b2    = (const float*)d_in[6];
    const float* Wout  = (const float*)d_in[7];
    const float* bout  = (const float*)d_in[8];
    float* out = (float*)d_out;

    const int* srcIdx = ei;
    const int* dstIdx = ei + NEDGES;

    // workspace carve-up (256B aligned)
    char* p = (char*)d_ws;
    auto carve = [&](size_t bytes) -> char* {
        char* r = p; p += (bytes + 255) & ~(size_t)255; return r;
    };
    float*    h    = (float*)   carve((size_t)NNODES * HID * 4);  // gemm outputs (reused both layers)
    float*    agg  = (float*)   carve((size_t)NNODES * HID * 4);  // scatter accumulator
    _Float16* h1r  = (_Float16*)carve((size_t)NNODES * HID * 2);  // relu(layer1) in f16
    float*    deg  = (float*)   carve((size_t)NNODES * 4);
    float*    dinv = (float*)   carve((size_t)NNODES * 4);
    float*    pool = (float*)   carve((size_t)NGRAPHS * HID * 4);
    float*    cnt  = (float*)   carve((size_t)NGRAPHS * 4);

    const int nb_nodes = (NNODES + 255) / 256;
    const int nb_edges = (NEDGES + 255) / 256;
    const int nb_gemm  = (NNODES + 127) / 128;
    const int nb_feat  = (NNODES * HID + 255) / 256;
    const int nb_scat  = (int)(((size_t)NEDGES * 32 + 255) / 256);

    // degrees + normalization
    init_deg_kernel <<<nb_nodes, 256, 0, stream>>>(deg, NNODES);
    count_deg_kernel<<<nb_edges, 256, 0, stream>>>(dstIdx, deg, NEDGES);
    dinv_kernel     <<<nb_nodes, 256, 0, stream>>>(deg, dinv, NNODES);

    // layer 1
    gemm_wmma_kernel<FIN, float><<<nb_gemm, 256, 0, stream>>>(x, W1, h, NNODES);
    hipMemsetAsync(agg, 0, (size_t)NNODES * HID * 4, stream);
    scatter_kernel<<<nb_scat, 256, 0, stream>>>(srcIdx, dstIdx, dinv, h, agg, NEDGES);
    finish1_kernel<<<nb_feat, 256, 0, stream>>>(agg, h, deg, b1, h1r, NNODES);

    // layer 2
    gemm_wmma_kernel<HID, _Float16><<<nb_gemm, 256, 0, stream>>>(h1r, W2, h, NNODES);
    hipMemsetAsync(agg, 0, (size_t)NNODES * HID * 4, stream);
    scatter_kernel<<<nb_scat, 256, 0, stream>>>(srcIdx, dstIdx, dinv, h, agg, NEDGES);

    // pooled mean + head
    hipMemsetAsync(pool, 0, (size_t)NGRAPHS * HID * 4, stream);
    hipMemsetAsync(cnt,  0, (size_t)NGRAPHS * 4, stream);
    finish2_pool_kernel<<<nb_feat, 256, 0, stream>>>(agg, h, deg, b2, batch, pool, cnt, NNODES);
    final_kernel<<<1, 128, 0, stream>>>(pool, cnt, Wout, bout, out);
}
